// TransformerBlock_41927470743716
// MI455X (gfx1250) — matrix operand order repaired and checked by reading, not compiled, not run on hardware
//
#include <hip/hip_runtime.h>

// ---------------------------------------------------------------------------
// Types for CDNA5 WMMA bf16 (v_wmma_f32_16x16x32_bf16)
// ---------------------------------------------------------------------------
typedef __attribute__((ext_vector_type(16))) __bf16 v16bf;
typedef __attribute__((ext_vector_type(8)))  __bf16 v8bf;
typedef __attribute__((ext_vector_type(4)))  __bf16 v4bf;
typedef __attribute__((ext_vector_type(8)))  float  v8f;
typedef __attribute__((ext_vector_type(4)))  int    v4i;

// address-space qualified element types for the async-to-LDS builtin
typedef __attribute__((address_space(1))) v4i v4i_glob;
typedef __attribute__((address_space(3))) v4i v4i_lds;

#define DEV static __device__ __forceinline__

DEV v8f vzero8() { v8f z = {0.f,0.f,0.f,0.f,0.f,0.f,0.f,0.f}; return z; }

DEV v8f wmma_bf16(v16bf a, v16bf b, v8f c) {
  return __builtin_amdgcn_wmma_f32_16x16x32_bf16(false, a, false, b, (short)0, c,
                                                 false, false);
}

// A-fragment (16x32 bf16), row-major source base[row][k], leading dim ld.
DEV v16bf load_frag_a(const __bf16* __restrict__ base, int ld, int row0, int k0,
                      int lane) {
  int r  = row0 + (lane & 15);
  int kb = k0 + ((lane & 16) ? 8 : 0);
  const __bf16* p = base + (size_t)r * ld + kb;
  v8bf lo = *(const v8bf*)(p);
  v8bf hi = *(const v8bf*)(p + 16);
  v16bf out;
#pragma unroll
  for (int i = 0; i < 8; ++i) { out[i] = lo[i]; out[i + 8] = hi[i]; }
  return out;
}

// B-fragment (32x16 bf16) from a K-contiguous (transposed) source baseT[col][k].
DEV v16bf load_frag_b(const __bf16* __restrict__ baseT, int ld, int k0, int col0,
                      int lane) {
  int c  = col0 + (lane & 15);
  int kb = k0 + ((lane & 16) ? 8 : 0);
  const __bf16* p = baseT + (size_t)c * ld + kb;
  v8bf lo = *(const v8bf*)(p);
  v8bf hi = *(const v8bf*)(p + 16);
  v16bf out;
#pragma unroll
  for (int i = 0; i < 8; ++i) { out[i] = lo[i]; out[i + 8] = hi[i]; }
  return out;
}

DEV float half_add(float v) {  // reduce over the 16 lanes of a half-wave
  v += __shfl_xor(v, 1); v += __shfl_xor(v, 2);
  v += __shfl_xor(v, 4); v += __shfl_xor(v, 8);
  return v;
}
DEV float half_max(float v) {
  v = fmaxf(v, __shfl_xor(v, 1)); v = fmaxf(v, __shfl_xor(v, 2));
  v = fmaxf(v, __shfl_xor(v, 4)); v = fmaxf(v, __shfl_xor(v, 8));
  return v;
}

// ---------------------------------------------------------------------------
// gfx1250 async global->LDS copy (16B granules), with safe fallback.
// Confirmed param 0 type: int4 addrspace(1)* (non-const).
// ---------------------------------------------------------------------------
#if defined(__gfx1250__) && __has_builtin(__builtin_amdgcn_global_load_async_to_lds_b128)
#define ATTN_ASYNC_STAGE 1
#else
#define ATTN_ASYNC_STAGE 0
#endif

DEV void async_cp16(void* lds_dst, const void* gsrc) {
#if ATTN_ASYNC_STAGE
  __builtin_amdgcn_global_load_async_to_lds_b128(
      (v4i_glob*)(void*)gsrc, (v4i_lds*)lds_dst, 0, 0);
#else
  *(v8bf*)lds_dst = *(const v8bf*)gsrc;
#endif
}

DEV void async_cp_wait() {
#if ATTN_ASYNC_STAGE
#if __has_builtin(__builtin_amdgcn_s_wait_asynccnt)
  __builtin_amdgcn_s_wait_asynccnt(0);
#else
  asm volatile("s_wait_asynccnt 0" ::: "memory");
#endif
#endif
}

// ---------------------------------------------------------------------------
// Weight convert + transpose: WT[n*K + k] = (bf16) W[k*N + n]
// ---------------------------------------------------------------------------
__global__ void wt_bf16_kernel(const float* __restrict__ W, __bf16* __restrict__ WT,
                               int K, int N) {
  int i = blockIdx.x * 256 + threadIdx.x;
  if (i >= K * N) return;
  int k = i / N;
  int n = i - k * N;
  WT[(size_t)n * K + k] = (__bf16)W[i];
}

// ---------------------------------------------------------------------------
// Shared LayerNorm over D=128, output bf16. One wave per row, 4 elems/lane.
// ---------------------------------------------------------------------------
__global__ void ln_bf16_kernel(const float* __restrict__ in, __bf16* __restrict__ out,
                               const float* __restrict__ g, const float* __restrict__ b,
                               int rows) {
  int lane = threadIdx.x & 31;
  int wv   = threadIdx.x >> 5;
  int row  = blockIdx.x * 8 + wv;
  if (row >= rows) return;
  float4 v = *(const float4*)(in + (size_t)row * 128 + lane * 4);
  float s = v.x + v.y + v.z + v.w;
#pragma unroll
  for (int m = 16; m >= 1; m >>= 1) s += __shfl_xor(s, m);
  float mu = s * (1.0f / 128.0f);
  float dx = v.x - mu, dy = v.y - mu, dz = v.z - mu, dw = v.w - mu;
  float q = dx * dx + dy * dy + dz * dz + dw * dw;
#pragma unroll
  for (int m = 16; m >= 1; m >>= 1) q += __shfl_xor(q, m);
  float rs = rsqrtf(q * (1.0f / 128.0f) + 1e-5f);
  float4 gv = *(const float4*)(g + lane * 4);
  float4 bv = *(const float4*)(b + lane * 4);
  v4bf o;
  o[0] = (__bf16)(dx * rs * gv.x + bv.x);
  o[1] = (__bf16)(dy * rs * gv.y + bv.y);
  o[2] = (__bf16)(dz * rs * gv.z + bv.z);
  o[3] = (__bf16)(dw * rs * gv.w + bv.w);
  *(v4bf*)(out + (size_t)row * 128 + lane * 4) = o;
}

// ---------------------------------------------------------------------------
// Generic bf16 WMMA GEMM: C[16 x 64] tile per wave. A row-major, WT transposed.
// mode 0: store bf16 row-major            (Q, K)
// mode 1: store bf16 TRANSPOSED [n][row]  (V -> VT)
// mode 2: relu(acc + bias) -> bf16        (MLP hidden)
// mode 3: acc + bias + resid -> f32       (final output)
// ---------------------------------------------------------------------------
__global__ void gemm_bf16_kernel(const __bf16* __restrict__ A,
                                 const __bf16* __restrict__ WT,
                                 const float* __restrict__ bias,
                                 const float* __restrict__ resid,
                                 void* __restrict__ Cout,
                                 int Kdim, int Ncols, int Rtot, int mode) {
  const int lane = threadIdx.x & 31;
  const int row0 = blockIdx.x * 16;
  const int col0 = blockIdx.y * 64;
  const int half = lane >> 4;
  const int nn   = lane & 15;

  v8f acc[4];
#pragma unroll
  for (int j = 0; j < 4; ++j) acc[j] = vzero8();

  for (int kk = 0; kk < Kdim; kk += 32) {
    v16bf a = load_frag_a(A, Kdim, row0, kk, lane);
#pragma unroll
    for (int j = 0; j < 4; ++j) {
      v16bf b = load_frag_b(WT, Kdim, kk, col0 + 16 * j, lane);
      acc[j] = wmma_bf16(a, b, acc[j]);
    }
  }

  if (mode == 0) {
    __bf16* C = (__bf16*)Cout;
#pragma unroll
    for (int j = 0; j < 4; ++j) {
      int n = col0 + 16 * j + nn;
#pragma unroll
      for (int r = 0; r < 8; ++r)
        C[(size_t)(row0 + r + 8 * half) * Ncols + n] = (__bf16)acc[j][r];
    }
  } else if (mode == 1) {
    __bf16* C = (__bf16*)Cout;
#pragma unroll
    for (int j = 0; j < 4; ++j) {
      v8bf pk;
#pragma unroll
      for (int r = 0; r < 8; ++r) pk[r] = (__bf16)acc[j][r];
      int n = col0 + 16 * j + nn;
      *(v8bf*)(C + (size_t)n * Rtot + row0 + 8 * half) = pk;
    }
  } else if (mode == 2) {
    __bf16* C = (__bf16*)Cout;
#pragma unroll
    for (int j = 0; j < 4; ++j) {
      int n = col0 + 16 * j + nn;
      float bb = bias[n];
#pragma unroll
      for (int r = 0; r < 8; ++r)
        C[(size_t)(row0 + r + 8 * half) * Ncols + n] =
            (__bf16)fmaxf(acc[j][r] + bb, 0.0f);
    }
  } else {
    float* C = (float*)Cout;
#pragma unroll
    for (int j = 0; j < 4; ++j) {
      int n = col0 + 16 * j + nn;
      float bb = bias[n];
#pragma unroll
      for (int r = 0; r < 8; ++r) {
        int row = row0 + r + 8 * half;
        C[(size_t)row * Ncols + n] =
            acc[j][r] + bb + resid[(size_t)row * Ncols + n];
      }
    }
  }
}

// ---------------------------------------------------------------------------
// Flash attention with smooth-softmax + residual + fused LayerNorm.
// 4 waves per block; each wave owns 16 query rows. 32 keys per step.
// K/V chunks are cooperatively staged to LDS (async-to-LDS on gfx1250)
// and shared by all 4 waves; P transpose is staged per-wave in LDS.
//   smooth_softmax(S) @ V = (0.1*relu(S)@V + exp(S-m)@V / Z) / (0.1*sumRelu + 1)
// Outputs: z1r = att + x  (f32), z2 = LN(z1r) (bf16)
// ---------------------------------------------------------------------------
__global__ __launch_bounds__(128) void attn_kernel(
    const __bf16* __restrict__ Q, const __bf16* __restrict__ Kmat,
    const __bf16* __restrict__ VT, const float* __restrict__ x,
    const float* __restrict__ g, const float* __restrict__ bvec,
    float* __restrict__ z1r, __bf16* __restrict__ z2, int Mtot) {
  __shared__ __bf16 Ks[32 * 128];          // [key_local][d]   8 KB
  __shared__ __bf16 Vs[128 * 32];          // [d][key_local]   8 KB
  __shared__ __bf16 ldsP[4][2][16 * 32];   // per-wave P (exp, relu) 8 KB

  const int tid  = threadIdx.x;
  const int lane = tid & 31;
  const int wid  = tid >> 5;
  const int row0 = blockIdx.x * 64 + wid * 16;
  const int half = lane >> 4;
  const int nn   = lane & 15;
  const float inv_sqrt_d = 0.08838834764831845f;  // 1/sqrt(128)

  // Q fragments for this wave's 16 rows x 128 dims
  v16bf qf[4];
#pragma unroll
  for (int d = 0; d < 4; ++d) qf[d] = load_frag_a(Q, 128, row0, 32 * d, lane);

  v8f eacc[8], racc[8];
#pragma unroll
  for (int c = 0; c < 8; ++c) { eacc[c] = vzero8(); racc[c] = vzero8(); }

  float mrow[8], le[8], rsum[8];
#pragma unroll
  for (int r = 0; r < 8; ++r) { mrow[r] = -1.0e30f; le[r] = 0.f; rsum[r] = 0.f; }

  for (int kb = 0; kb < Mtot; kb += 32) {
    if (kb + 32 < Mtot)
      __builtin_prefetch(Kmat + (size_t)(kb + 32) * 128, 0, 1);

    // ---- cooperative stage: K chunk (contiguous 8KB) + VT chunk (128x32) ----
    {
      const char* ksrc = (const char*)Kmat + (size_t)kb * 256;
      char* kdst = (char*)Ks;
#pragma unroll
      for (int i = 0; i < 4; ++i) {
        int off = tid * 16 + i * 2048;
        async_cp16(kdst + off, ksrc + off);
      }
      const char* vsrc = (const char*)(VT + (size_t)tid * Mtot + kb);
      char* vdst = (char*)(Vs + tid * 32);
#pragma unroll
      for (int i = 0; i < 4; ++i)
        async_cp16(vdst + i * 16, vsrc + i * 16);
      async_cp_wait();
    }
    __syncthreads();

    // ---- S tile: 16 rows x 32 keys, contraction over D=128 (from LDS K) ----
    v8f s[2];
    s[0] = vzero8(); s[1] = vzero8();
#pragma unroll
    for (int d = 0; d < 4; ++d) {
#pragma unroll
      for (int j = 0; j < 2; ++j) {
        v16bf b = load_frag_b(Ks, 128, 32 * d, 16 * j, lane);
        s[j] = wmma_bf16(qf[d], b, s[j]);
      }
    }
#pragma unroll
    for (int j = 0; j < 2; ++j)
#pragma unroll
      for (int r = 0; r < 8; ++r) s[j][r] *= inv_sqrt_d;

    // ---- online max per row ----
    float mnew[8], cs[8];
#pragma unroll
    for (int r = 0; r < 8; ++r) {
      float tm = half_max(fmaxf(s[0][r], s[1][r]));
      mnew[r] = fmaxf(mrow[r], tm);
      cs[r]   = __expf(mrow[r] - mnew[r]);
      mrow[r] = mnew[r];
    }
#pragma unroll
    for (int c = 0; c < 8; ++c)
#pragma unroll
      for (int r = 0; r < 8; ++r) eacc[c][r] *= cs[r];
#pragma unroll
    for (int r = 0; r < 8; ++r) le[r] *= cs[r];

    // ---- P = exp(S-m), Prelu = relu(S); row sums; stage for transpose ----
    float es[8], rls[8];
#pragma unroll
    for (int r = 0; r < 8; ++r) { es[r] = 0.f; rls[r] = 0.f; }
#pragma unroll
    for (int j = 0; j < 2; ++j) {
#pragma unroll
      for (int r = 0; r < 8; ++r) {
        float sv = s[j][r];
        float pe = __expf(sv - mrow[r]);
        float pr = fmaxf(sv, 0.0f);
        es[r]  += pe;
        rls[r] += pr;
        int row = r + 8 * half;
        int col = 16 * j + nn;
        ldsP[wid][0][row * 32 + col] = (__bf16)pe;
        ldsP[wid][1][row * 32 + col] = (__bf16)pr;
      }
    }
#pragma unroll
    for (int r = 0; r < 8; ++r) {
      le[r]   += half_add(es[r]);
      rsum[r] += half_add(rls[r]);
    }
    __syncthreads();

    // ---- reload P tiles in A-fragment layout; accumulate P@V from LDS V ----
    v16bf pa  = load_frag_a(&ldsP[wid][0][0], 32, 0, 0, lane);
    v16bf pra = load_frag_a(&ldsP[wid][1][0], 32, 0, 0, lane);
#pragma unroll
    for (int c = 0; c < 8; ++c) {
      v16bf vb = load_frag_b(Vs, 32, 0, 16 * c, lane);
      eacc[c] = wmma_bf16(pa,  vb, eacc[c]);
      racc[c] = wmma_bf16(pra, vb, racc[c]);
    }
    __syncthreads();  // protect Ks/Vs before next stage
  }

  // ---- finalize: att, residual, fused LayerNorm over this wave's 16 rows ----
  float zf[8][8];
#pragma unroll
  for (int c = 0; c < 8; ++c) {
#pragma unroll
    for (int r = 0; r < 8; ++r) {
      float att = (0.1f * racc[c][r] + eacc[c][r] / le[r]) /
                  (0.1f * rsum[r] + 1.0f);
      int row = row0 + r + 8 * half;
      int col = 16 * c + nn;
      float z = att + x[(size_t)row * 128 + col];
      z1r[(size_t)row * 128 + col] = z;
      zf[c][r] = z;
    }
  }
  float mu[8], rsn[8];
#pragma unroll
  for (int r = 0; r < 8; ++r) {
    float t = 0.f;
#pragma unroll
    for (int c = 0; c < 8; ++c) t += zf[c][r];
    mu[r] = half_add(t) * (1.0f / 128.0f);
  }
#pragma unroll
  for (int r = 0; r < 8; ++r) {
    float t = 0.f;
#pragma unroll
    for (int c = 0; c < 8; ++c) {
      float d = zf[c][r] - mu[r];
      t += d * d;
    }
    rsn[r] = rsqrtf(half_add(t) * (1.0f / 128.0f) + 1e-5f);
  }
#pragma unroll
  for (int c = 0; c < 8; ++c) {
    int col = 16 * c + nn;
    float gv = g[col];
    float bv = bvec[col];
#pragma unroll
    for (int r = 0; r < 8; ++r) {
      int row = row0 + r + 8 * half;
      z2[(size_t)row * 128 + col] = (__bf16)((zf[c][r] - mu[r]) * rsn[r] * gv + bv);
    }
  }
}

// ---------------------------------------------------------------------------
// Host-side pipeline
// ---------------------------------------------------------------------------
extern "C" void kernel_launch(void* const* d_in, const int* in_sizes, int n_in,
                              void* d_out, int out_size, void* d_ws, size_t ws_size,
                              hipStream_t stream) {
  (void)in_sizes; (void)n_in; (void)out_size; (void)ws_size;
  const int N = 8192, M = 8192, D = 128, H = 512;

  const float* x    = (const float*)d_in[0];
  const float* y    = (const float*)d_in[1];
  const float* Wq   = (const float*)d_in[2];
  const float* Wk   = (const float*)d_in[3];
  const float* Wv   = (const float*)d_in[4];
  const float* ln_g = (const float*)d_in[5];
  const float* ln_b = (const float*)d_in[6];
  const float* W1   = (const float*)d_in[7];
  const float* b1   = (const float*)d_in[8];
  const float* W2   = (const float*)d_in[9];
  const float* b2   = (const float*)d_in[10];
  const float* W3   = (const float*)d_in[11];
  const float* b3   = (const float*)d_in[12];

  char* ws = (char*)d_ws;
  size_t off = 0;
  auto alloc = [&](size_t bytes) { char* p = ws + off; off += (bytes + 255) & ~(size_t)255; return p; };
  __bf16* Xn  = (__bf16*)alloc((size_t)N * D * 2);
  __bf16* Wn  = (__bf16*)alloc((size_t)M * D * 2);
  __bf16* Qb  = (__bf16*)alloc((size_t)N * D * 2);
  __bf16* Kb  = (__bf16*)alloc((size_t)M * D * 2);
  __bf16* VTb = (__bf16*)alloc((size_t)D * M * 2);
  float*  z1r = (float*) alloc((size_t)N * D * 4);
  __bf16* z2  = (__bf16*)alloc((size_t)N * D * 2);
  __bf16* h1  = (__bf16*)alloc((size_t)N * H * 2);
  __bf16* h2  = (__bf16*)alloc((size_t)N * H * 2);
  __bf16* WqT = (__bf16*)alloc((size_t)D * D * 2);
  __bf16* WkT = (__bf16*)alloc((size_t)D * D * 2);
  __bf16* WvT = (__bf16*)alloc((size_t)D * D * 2);
  __bf16* W1T = (__bf16*)alloc((size_t)D * H * 2);
  __bf16* W2T = (__bf16*)alloc((size_t)H * H * 2);
  __bf16* W3T = (__bf16*)alloc((size_t)H * D * 2);

  // 1) weights -> bf16 transposed
  auto cvt = [&](const float* W, __bf16* WT, int K, int Ncol) {
    int tot = K * Ncol;
    wt_bf16_kernel<<<(tot + 255) / 256, 256, 0, stream>>>(W, WT, K, Ncol);
  };
  cvt(Wq, WqT, D, D); cvt(Wk, WkT, D, D); cvt(Wv, WvT, D, D);
  cvt(W1, W1T, D, H); cvt(W2, W2T, H, H); cvt(W3, W3T, H, D);

  // 2) shared LayerNorm on both streams
  ln_bf16_kernel<<<N / 8, 256, 0, stream>>>(x, Xn, ln_g, ln_b, N);
  ln_bf16_kernel<<<M / 8, 256, 0, stream>>>(y, Wn, ln_g, ln_b, M);

  // 3) projections: Q, K row-major; V stored transposed
  gemm_bf16_kernel<<<dim3(N / 16, D / 64), 32, 0, stream>>>(
      Xn, WqT, nullptr, nullptr, Qb, D, D, N, 0);
  gemm_bf16_kernel<<<dim3(M / 16, D / 64), 32, 0, stream>>>(
      Wn, WkT, nullptr, nullptr, Kb, D, D, M, 0);
  gemm_bf16_kernel<<<dim3(M / 16, D / 64), 32, 0, stream>>>(
      Wn, WvT, nullptr, nullptr, VTb, D, D, M, 1);

  // 4) flash attention (smooth softmax) + residual + fused LN
  attn_kernel<<<N / 64, 128, 0, stream>>>(Qb, Kb, VTb, x, ln_g, ln_b, z1r, z2, M);

  // 5) MLP
  gemm_bf16_kernel<<<dim3(N / 16, H / 64), 32, 0, stream>>>(
      z2, W1T, b1, nullptr, h1, D, H, N, 2);
  gemm_bf16_kernel<<<dim3(N / 16, H / 64), 32, 0, stream>>>(
      h1, W2T, b2, nullptr, h2, H, H, N, 2);
  gemm_bf16_kernel<<<dim3(N / 16, D / 64), 32, 0, stream>>>(
      h2, W3T, b3, z1r, d_out, H, D, N, 3);
}
